// MambaVSum_71244917506113
// MI455X (gfx1250) — matrix operand order repaired and checked by reading, not compiled, not run on hardware
//
#include <hip/hip_runtime.h>
#include <math.h>

// ---------------------------------------------------------------------------
// CDNA5 / gfx1250: wave32, WMMA 16x16x32 bf16 -> f32.
// GEMM: 64x128 block tile, 8 waves x (32x32 out, 4 accumulators, 4 WMMA/step).
// Weights pre-converted once per launch to bf16 transposed K-padded layout, so
// B staging is pure b128 copy; A staged f32->bf16 via v_cvt_pk_bf16_f32.
// Register-staged software pipeline overlaps global loads with WMMA compute.
// ---------------------------------------------------------------------------

typedef __attribute__((ext_vector_type(16))) __bf16 v16bf;
typedef __attribute__((ext_vector_type(8)))  __bf16 v8bf;
typedef __attribute__((ext_vector_type(4)))  __bf16 v4bf;
typedef __attribute__((ext_vector_type(8)))  float  v8f;

#define ACT_NONE     0
#define ACT_SIGMOID  1
#define ACT_RELU     2
#define ACT_SOFTPLUS 3

__device__ __forceinline__ float sigmoidf_(float x) { return 1.f / (1.f + __expf(-x)); }
__device__ __forceinline__ float siluf_(float x)    { return x * sigmoidf_(x); }
__device__ __forceinline__ float act_apply(float x, int act) {
  if (act == ACT_SIGMOID)  return sigmoidf_(x);
  if (act == ACT_RELU)     return fmaxf(x, 0.f);
  if (act == ACT_SOFTPLUS) return (x > 20.f) ? x : __logf(1.f + __expf(x));
  return x;
}

#define CAT16(a, b) __builtin_shufflevector(a, b, 0,1,2,3,4,5,6,7,8,9,10,11,12,13,14,15)

// weight pre-conversion: w[K x N] f32 -> wt[N x Kp] bf16 (transposed, zero-pad K..Kp)
__global__ void k_wT_bf16(const float* __restrict__ w, __bf16* __restrict__ wt,
                          int K, int N, int Kp)
{
  int idx = blockIdx.x * 256 + threadIdx.x;
  if (idx >= N * Kp) return;
  int n = idx / Kp, k = idx - n * Kp;
  wt[idx] = (__bf16)((k < K) ? w[(size_t)k * N + n] : 0.f);
}

// ---------------- WMMA GEMM: C = act(A[MxK] @ B[KxN] + bias) ----------------
#define TM 64
#define TN 128
#define TK 32
#define SAS 40   // LDS row strides (bf16): 80B, 16B-aligned, conflict-free b128 reads
#define SBS 40

// fragment gather + 4x WMMA on the currently staged tile
#define FRAG_COMPUTE()                                                          \
  do {                                                                          \
    v16bf fb0, fb1, fa0, fa1;                                                   \
    { const __bf16* p = &sBt[wn * 32 + l16][16 * half];                         \
      fb0 = CAT16(*(const v8bf*)p, *(const v8bf*)(p + 8)); }                    \
    { const __bf16* p = &sBt[wn * 32 + 16 + l16][16 * half];                    \
      fb1 = CAT16(*(const v8bf*)p, *(const v8bf*)(p + 8)); }                    \
    { const __bf16* p = &sA[wm * 32 + l16][16 * half];                          \
      fa0 = CAT16(*(const v8bf*)p, *(const v8bf*)(p + 8)); }                    \
    { const __bf16* p = &sA[wm * 32 + 16 + l16][16 * half];                     \
      fa1 = CAT16(*(const v8bf*)p, *(const v8bf*)(p + 8)); }                    \
    acc00 = __builtin_amdgcn_wmma_f32_16x16x32_bf16(false, fa0, false, fb0,     \
                                                    (short)0, acc00, false, false); \
    acc01 = __builtin_amdgcn_wmma_f32_16x16x32_bf16(false, fa0, false, fb1,     \
                                                    (short)0, acc01, false, false); \
    acc10 = __builtin_amdgcn_wmma_f32_16x16x32_bf16(false, fa1, false, fb0,     \
                                                    (short)0, acc10, false, false); \
    acc11 = __builtin_amdgcn_wmma_f32_16x16x32_bf16(false, fa1, false, fb1,     \
                                                    (short)0, acc11, false, false); \
  } while (0)

__global__ __launch_bounds__(256)
void k_gemm_bf16(const float* __restrict__ A, const __bf16* __restrict__ Bt,
                 const float* __restrict__ bias, float* __restrict__ C,
                 int M, int N, int Kp, int lda, int ldbt, int ldc, int act)
{
  __shared__ __bf16 sA [TM][SAS];   // A tile (row = M, col = K)
  __shared__ __bf16 sBt[TN][SBS];   // B tile transposed (row = N, col = K)

  const int tid  = threadIdx.x;     // 256 threads = 8 waves (wave32)
  const int lane = tid & 31;
  const int wave = tid >> 5;
  const int wm   = wave >> 2;       // 0..1 : 32-row slab
  const int wn   = wave & 3;        // 0..3 : 32-col slab
  const int half = lane >> 4;
  const int l16  = lane & 15;
  const int m0   = blockIdx.y * TM;
  const int n0   = blockIdx.x * TN;

  v8f acc00 = {0.f,0.f,0.f,0.f,0.f,0.f,0.f,0.f};
  v8f acc01 = acc00, acc10 = acc00, acc11 = acc00;

  // staging coordinates (per thread)
  const int rA = tid >> 3, cA = (tid & 7) * 4;    // A: float4; rows rA, rA+32
  const int rB = tid >> 2, kB = (tid & 3) * 8;    // B: v8bf;  rows rB, rB+64

  if ((m0 + TM <= M) && (n0 + TN <= N)) {
    // ---------------- fast path: register-staged pipeline ----------------
    const float*  pa0 = A  + (size_t)(m0 + rA)      * lda  + cA;
    const float*  pa1 = A  + (size_t)(m0 + rA + 32) * lda  + cA;
    const __bf16* pb0 = Bt + (size_t)(n0 + rB)      * ldbt + kB;
    const __bf16* pb1 = Bt + (size_t)(n0 + rB + 64) * ldbt + kB;
    float4 ra0 = *(const float4*)pa0;
    float4 ra1 = *(const float4*)pa1;
    v8bf   rb0 = *(const v8bf*)pb0;
    v8bf   rb1 = *(const v8bf*)pb1;

    for (int k0 = 0; k0 < Kp; k0 += TK) {
      // commit staged registers to LDS
      { v4bf h; h[0] = (__bf16)ra0.x; h[1] = (__bf16)ra0.y;
                h[2] = (__bf16)ra0.z; h[3] = (__bf16)ra0.w;
        *(v4bf*)&sA[rA][cA] = h; }
      { v4bf h; h[0] = (__bf16)ra1.x; h[1] = (__bf16)ra1.y;
                h[2] = (__bf16)ra1.z; h[3] = (__bf16)ra1.w;
        *(v4bf*)&sA[rA + 32][cA] = h; }
      *(v8bf*)&sBt[rB][kB]      = rb0;
      *(v8bf*)&sBt[rB + 64][kB] = rb1;
      __syncthreads();

      // issue next tile's global loads; they complete while WMMAs run
      if (k0 + TK < Kp) {
        pa0 += TK; pa1 += TK; pb0 += TK; pb1 += TK;
        ra0 = *(const float4*)pa0;
        ra1 = *(const float4*)pa1;
        rb0 = *(const v8bf*)pb0;
        rb1 = *(const v8bf*)pb1;
      }
      FRAG_COMPUTE();
      __syncthreads();
    }
  } else {
    // ---------------- slow path: guarded scalar staging ----------------
    for (int k0 = 0; k0 < Kp; k0 += TK) {
      #pragma unroll
      for (int i = 0; i < 8; ++i) {                 // A: 64x32 elems
        int idx = tid + i * 256;
        int r = idx >> 5, c = idx & 31;
        int gr = m0 + r;
        sA[r][c] = (__bf16)((gr < M) ? A[(size_t)gr * lda + k0 + c] : 0.f);
      }
      #pragma unroll
      for (int i = 0; i < 16; ++i) {                // B: 128x32 elems (bf16 src)
        int idx = tid + i * 256;
        int r = idx >> 5, c = idx & 31;             // r = N row of Bt, c = K col
        int gn = n0 + r;
        sBt[r][c] = (gn < N) ? Bt[(size_t)gn * ldbt + k0 + c] : (__bf16)0.f;
      }
      __syncthreads();
      FRAG_COMPUTE();
      __syncthreads();
    }
  }

  // C/D layout: elem j -> row j + 8*half (within 16), col l16.
  const int cn0 = n0 + wn * 32 + l16;
  const int cn1 = cn0 + 16;
  const int rm0 = m0 + wm * 32;
  #pragma unroll
  for (int j = 0; j < 8; ++j) {
    int cm = rm0 + j + 8 * half;
    if (cm < M) {
      if (cn0 < N) C[(size_t)cm * ldc + cn0] =
          act_apply(acc00[j] + (bias ? bias[cn0] : 0.f), act);
      if (cn1 < N) C[(size_t)cm * ldc + cn1] =
          act_apply(acc01[j] + (bias ? bias[cn1] : 0.f), act);
    }
    int cm1 = rm0 + 16 + j + 8 * half;
    if (cm1 < M) {
      if (cn0 < N) C[(size_t)cm1 * ldc + cn0] =
          act_apply(acc10[j] + (bias ? bias[cn0] : 0.f), act);
      if (cn1 < N) C[(size_t)cm1 * ldc + cn1] =
          act_apply(acc11[j] + (bias ? bias[cn1] : 0.f), act);
    }
  }
}

// ---------------- LayerNorm family (blockDim == d, d in {128,256}) ----------
__global__ void k_ln(const float* __restrict__ in, const float* __restrict__ g,
                     const float* __restrict__ b, float* __restrict__ out, int d)
{
  __shared__ float red[256];
  const int row = blockIdx.x, t = threadIdx.x;
  float v = in[(size_t)row * d + t];
  red[t] = v; __syncthreads();
  for (int s = d >> 1; s > 0; s >>= 1) { if (t < s) red[t] += red[t + s]; __syncthreads(); }
  float mean = red[0] / d; __syncthreads();
  float dv = v - mean;
  red[t] = dv * dv; __syncthreads();
  for (int s = d >> 1; s > 0; s >>= 1) { if (t < s) red[t] += red[t + s]; __syncthreads(); }
  float inv = rsqrtf(red[0] / d + 1e-5f);
  out[(size_t)row * d + t] = dv * inv * g[t] + b[t];
}

// x = LN(base? + gate*f + (1-gate)*bw[rev]) — fusion gate and layer merge.
__global__ void k_gatecomb_ln(const float* __restrict__ base, const float* __restrict__ f,
                              const float* __restrict__ bw, const float* __restrict__ gate,
                              const float* __restrict__ g, const float* __restrict__ b,
                              float* __restrict__ out, int N, int d, int revb)
{
  __shared__ float red[256];
  const int row = blockIdx.x, t = threadIdx.x;
  const int br = revb ? (N - 1 - row) : row;
  float fv = f[(size_t)row * d + t];
  float bv = bw[(size_t)br * d + t];
  float gv = gate[(size_t)row * d + t];
  float v = gv * fv + (1.f - gv) * bv;
  if (base) v += base[(size_t)row * d + t];
  red[t] = v; __syncthreads();
  for (int s = d >> 1; s > 0; s >>= 1) { if (t < s) red[t] += red[t + s]; __syncthreads(); }
  float mean = red[0] / d; __syncthreads();
  float dv = v - mean;
  red[t] = dv * dv; __syncthreads();
  for (int s = d >> 1; s > 0; s >>= 1) { if (t < s) red[t] += red[t + s]; __syncthreads(); }
  float inv = rsqrtf(red[0] / d + 1e-5f);
  out[(size_t)row * d + t] = dv * inv * g[t] + b[t];
}

__global__ void k_res_ln(const float* __restrict__ a, const float* __restrict__ c,
                         const float* __restrict__ g, const float* __restrict__ b,
                         float* __restrict__ out, int d)
{
  __shared__ float red[256];
  const int row = blockIdx.x, t = threadIdx.x;
  float v = a[(size_t)row * d + t] + c[(size_t)row * d + t];
  red[t] = v; __syncthreads();
  for (int s = d >> 1; s > 0; s >>= 1) { if (t < s) red[t] += red[t + s]; __syncthreads(); }
  float mean = red[0] / d; __syncthreads();
  float dv = v - mean;
  red[t] = dv * dv; __syncthreads();
  for (int s = d >> 1; s > 0; s >>= 1) { if (t < s) red[t] += red[t + s]; __syncthreads(); }
  float inv = rsqrtf(red[0] / d + 1e-5f);
  out[(size_t)row * d + t] = dv * inv * g[t] + b[t];
}

// ---------------- elementwise helpers ----------------
__global__ void k_concat2(float* __restrict__ dst, const float* __restrict__ a,
                          const float* __restrict__ b, int N, int da, int db, int revb)
{
  int idx = blockIdx.x * 256 + threadIdx.x;
  int dt = da + db;
  if (idx >= N * dt) return;
  int row = idx / dt, col = idx - row * dt;
  if (col < da) dst[idx] = a[(size_t)row * da + col];
  else {
    int br = revb ? (N - 1 - row) : row;
    dst[idx] = b[(size_t)br * db + (col - da)];
  }
}

__global__ void k_reverse(float* __restrict__ out, const float* __restrict__ in, int N, int d)
{
  int idx = blockIdx.x * 256 + threadIdx.x;
  if (idx >= N * d) return;
  int r = idx / d, c = idx - r * d;
  out[idx] = in[(size_t)(N - 1 - r) * d + c];
}

// depthwise causal conv (dc taps) + SiLU; input = xz[:, :di], row stride 2*di
__global__ void k_conv_silu(const float* __restrict__ xz, const float* __restrict__ w,
                            const float* __restrict__ b, float* __restrict__ u,
                            int N, int di, int dc)
{
  int idx = blockIdx.x * 256 + threadIdx.x;
  if (idx >= N * di) return;
  int t = idx / di, c = idx - t * di;
  float acc = b[c];
  for (int k = 0; k < dc; ++k) {
    int tt = t + k - (dc - 1);
    if (tt >= 0) acc += xz[(size_t)tt * (2 * di) + c] * w[c * dc + k];
  }
  u[idx] = siluf_(acc);
}

// selective scan: one lane per channel, 16 states in VGPRs
__global__ void k_scan(const float* __restrict__ dbc,    // N x 48 (dt|B|C)
                       const float* __restrict__ delta,  // N x di
                       const float* __restrict__ u,      // N x di
                       const float* __restrict__ A_log,  // di x 16
                       const float* __restrict__ Dp,     // di
                       float* __restrict__ y,            // N x di
                       int N, int di)
{
  int c = blockIdx.x * blockDim.x + threadIdx.x;
  if (c >= di) return;
  float Arow[16], h[16];
  #pragma unroll
  for (int s = 0; s < 16; ++s) { Arow[s] = -__expf(A_log[c * 16 + s]); h[s] = 0.f; }
  float Dc = Dp[c];
  for (int t = 0; t < N; ++t) {
    float dt = delta[(size_t)t * di + c];
    float ut = u[(size_t)t * di + c];
    const float* Bc = dbc + (size_t)t * 48 + 16;
    const float* Cc = dbc + (size_t)t * 48 + 32;
    float out = 0.f;
    #pragma unroll
    for (int s = 0; s < 16; ++s) {
      float dA = __expf(dt * Arow[s]);
      h[s] = dA * h[s] + dt * Bc[s] * ut;
      out += h[s] * Cc[s];
    }
    y[(size_t)t * di + c] = out + Dc * ut;
  }
}

__global__ void k_ymul(float* __restrict__ y, const float* __restrict__ xz, int N, int di)
{
  int idx = blockIdx.x * 256 + threadIdx.x;
  if (idx >= N * di) return;
  int t = idx / di, c = idx - t * di;
  y[idx] *= siluf_(xz[(size_t)t * (2 * di) + di + c]);
}

// strided mean-pool + linear resample back to length N
__global__ void k_pool(const float* __restrict__ x, float* __restrict__ p, int N, int d, int s)
{
  int idx = blockIdx.x * 256 + threadIdx.x;
  if (idx >= N * d) return;
  int t = idx / d, j = idx - t * d;
  int Np = N / s;
  float pos = fmaxf((t + 0.5f) / s - 0.5f, 0.f);
  int lo = (int)floorf(pos);
  int hi = min(lo + 1, Np - 1);
  float fr = pos - (float)lo;
  float a = 0.f, b = 0.f;
  for (int q = 0; q < s; ++q) {
    a += x[(size_t)(lo * s + q) * d + j];
    b += x[(size_t)(hi * s + q) * d + j];
  }
  a /= s; b /= s;
  p[idx] = a * (1.f - fr) + b * fr;
}

__global__ void k_gather(float* __restrict__ dst, const float* __restrict__ src,
                         const int* __restrict__ idx, int K, int d)
{
  int i = blockIdx.x * 256 + threadIdx.x;
  if (i >= K * d) return;
  int r = i / d, c = i - r * d;
  dst[i] = src[(size_t)idx[r] * d + c];
}

// cross-attention, H=4 heads, hd=64, Kc keys; one block (128 threads) per query
__global__ void k_attn(const float* __restrict__ Q, const float* __restrict__ Kd,
                       const float* __restrict__ Vd, float* __restrict__ O, int Kc)
{
  __shared__ float sq[256];
  __shared__ float sc[4][128];
  __shared__ float red[128];
  const int n = blockIdx.x, t = threadIdx.x;   // t: 0..127
  sq[t]       = Q[(size_t)n * 256 + t];
  sq[t + 128] = Q[(size_t)n * 256 + 128 + t];
  __syncthreads();
  for (int h = 0; h < 4; ++h) {
    const float* krow = Kd + (size_t)t * 256 + h * 64;
    float s = 0.f;
    #pragma unroll 8
    for (int e = 0; e < 64; ++e) s += sq[h * 64 + e] * krow[e];
    sc[h][t] = s * 0.125f;  // 1/sqrt(64)
  }
  __syncthreads();
  for (int h = 0; h < 4; ++h) {
    float v = sc[h][t];
    red[t] = v; __syncthreads();
    for (int s = 64; s > 0; s >>= 1) { if (t < s) red[t] = fmaxf(red[t], red[t + s]); __syncthreads(); }
    float mx = red[0]; __syncthreads();
    float ex = __expf(v - mx);
    red[t] = ex; __syncthreads();
    for (int s = 64; s > 0; s >>= 1) { if (t < s) red[t] += red[t + s]; __syncthreads(); }
    float sum = red[0]; __syncthreads();
    sc[h][t] = ex / sum;
    __syncthreads();
  }
  for (int r = 0; r < 2; ++r) {
    int j = t + r * 128;             // output channel 0..255
    int h = j >> 6, e = j & 63;
    float acc = 0.f;
    for (int k = 0; k < Kc; ++k) acc += sc[h][k] * Vd[(size_t)k * 256 + h * 64 + e];
    O[(size_t)n * 256 + j] = acc;
  }
}

__global__ void k_score(const float* __restrict__ h, const float* __restrict__ w,
                        const float* __restrict__ b, float* __restrict__ out, int N, int dh)
{
  int n = blockIdx.x * blockDim.x + threadIdx.x;
  if (n >= N) return;
  float acc = b[0];
  for (int j = 0; j < dh; ++j) acc += h[(size_t)n * dh + j] * w[j];
  out[n] = sigmoidf_(acc);
}

// ---------------------------------------------------------------------------
extern "C" void kernel_launch(void* const* d_in, const int* in_sizes, int n_in,
                              void* d_out, int out_size, void* d_ws, size_t ws_size,
                              hipStream_t stream)
{
  const int N = 4096, Dv = 768, Da = 128, d = 256, di = 512, dc = 4, L = 4, dr = 16, Kc = 128;

  const float* visual    = (const float*)d_in[0];
  const float* audio     = (const float*)d_in[1];
  const int*   cp_idx    = (const int*)  d_in[2];
  const float* fus_wv    = (const float*)d_in[3];
  const float* fus_bv    = (const float*)d_in[4];
  const float* fus_wa    = (const float*)d_in[5];
  const float* fus_ba    = (const float*)d_in[6];
  const float* fus_wg    = (const float*)d_in[7];
  const float* fus_bg    = (const float*)d_in[8];
  const float* fus_ln_g  = (const float*)d_in[9];
  const float* fus_ln_b  = (const float*)d_in[10];
  const float* m_in_w    = (const float*)d_in[11];
  const float* m_conv_w  = (const float*)d_in[12];
  const float* m_conv_b  = (const float*)d_in[13];
  const float* m_xproj_w = (const float*)d_in[14];
  const float* m_dt_w    = (const float*)d_in[15];
  const float* m_dt_b    = (const float*)d_in[16];
  const float* m_Alog    = (const float*)d_in[17];
  const float* m_D       = (const float*)d_in[18];
  const float* m_out_w   = (const float*)d_in[19];
  const float* g_w       = (const float*)d_in[20];
  const float* g_b       = (const float*)d_in[21];
  const float* enc_ln_g  = (const float*)d_in[22];
  const float* enc_ln_b  = (const float*)d_in[23];
  const float* pool_w    = (const float*)d_in[24];
  const float* pool_b    = (const float*)d_in[25];
  const float* comb_w    = (const float*)d_in[26];
  const float* comb_b    = (const float*)d_in[27];
  const float* comb_ln_g = (const float*)d_in[28];
  const float* comb_ln_b = (const float*)d_in[29];
  const float* qkv_w     = (const float*)d_in[30];
  const float* qkv_b     = (const float*)d_in[31];
  const float* attn_out_w= (const float*)d_in[32];
  const float* attn_out_b= (const float*)d_in[33];
  const float* cp_ln_g   = (const float*)d_in[34];
  const float* cp_ln_b   = (const float*)d_in[35];
  const float* s1_w      = (const float*)d_in[36];
  const float* s1_b      = (const float*)d_in[37];
  const float* s_ln_g    = (const float*)d_in[38];
  const float* s_ln_b    = (const float*)d_in[39];
  const float* s2_w      = (const float*)d_in[40];
  const float* s2_b      = (const float*)d_in[41];

  // ---- float workspace (~81 MB, region reuse) ----
  float* W = (float*)d_ws;
  size_t off = 0;
  auto alloc = [&](size_t n) { float* p = W + off; off += n; return p; };
  float* X    = alloc((size_t)N * d);        // running hidden state / encoded
  float* XREV = alloc((size_t)N * d);        // reversed x for backward Mamba
  float* XZ   = alloc((size_t)N * 2 * di);   // mamba in-proj; reused for attention bufs
  float* U    = alloc((size_t)N * di);
  float* DEL  = alloc((size_t)N * di);       // + Y region reused as P3 (N x 3d) later
  float* Y    = alloc((size_t)N * di);
  float* DBC  = alloc((size_t)N * 48);
  float* F    = alloc((size_t)N * d);
  float* BWD  = alloc((size_t)N * d);
  float* CAT  = alloc((size_t)N * 2 * d);
  float* T1   = alloc((size_t)N * d);        // v ; later xp
  float* T2   = alloc((size_t)N * d);        // a ; later pool scratch
  float* T3   = alloc((size_t)N * d);        // gates / pre-LN temps
  (void)ws_size; (void)n_in; (void)in_sizes; (void)out_size;

  // ---- bf16 transposed weight arena (~10 MB), 32B aligned ----
  off = (off + 7) & ~(size_t)7;
  __bf16* warena = (__bf16*)(W + off);
  size_t woff = 0;
  auto cvtw = [&](const float* w, int K, int Nn) {
    int Kp = (K + 31) & ~31;
    __bf16* p = warena + woff;
    woff += (size_t)Nn * Kp;
    int tot = Nn * Kp;
    k_wT_bf16<<<(tot + 255) / 256, 256, 0, stream>>>(w, p, K, Nn, Kp);
    return p;
  };

  const __bf16* wt_fus_wv = cvtw(fus_wv, Dv, d);
  const __bf16* wt_fus_wa = cvtw(fus_wa, Da, d);
  const __bf16* wt_fus_wg = cvtw(fus_wg, 2 * d, d);
  const __bf16* wt_in[8]; const __bf16* wt_xp[8];
  const __bf16* wt_dt[8]; const __bf16* wt_out[8];
  for (int s = 0; s < 8; ++s) {
    wt_in[s]  = cvtw(m_in_w    + (size_t)s * d * 2 * di, d,  2 * di);
    wt_xp[s]  = cvtw(m_xproj_w + (size_t)s * di * 48,    di, 48);
    wt_dt[s]  = cvtw(m_dt_w    + (size_t)s * dr * di,    dr, di);
    wt_out[s] = cvtw(m_out_w   + (size_t)s * di * d,     di, d);
  }
  const __bf16* wt_gw[4];
  for (int l = 0; l < L; ++l) wt_gw[l] = cvtw(g_w + (size_t)l * 2 * d * d, 2 * d, d);
  const __bf16* wt_pool[3];
  for (int i = 0; i < 3; ++i) wt_pool[i] = cvtw(pool_w + (size_t)i * d * d, d, d);
  const __bf16* wt_comb = cvtw(comb_w, 3 * d, d);
  const __bf16* wt_qkv[3];
  for (int i = 0; i < 3; ++i) wt_qkv[i] = cvtw(qkv_w + (size_t)i * d * d, d, d);
  const __bf16* wt_ao = cvtw(attn_out_w, d, d);
  const __bf16* wt_s1 = cvtw(s1_w, d, 128);

  auto gemm = [&](const float* A, int lda, const __bf16* Bt, const float* bias,
                  float* C, int ldc, int M, int Nn, int K, int act) {
    int Kp = (K + 31) & ~31;
    dim3 grid((Nn + TN - 1) / TN, (M + TM - 1) / TM);
    k_gemm_bf16<<<grid, 256, 0, stream>>>(A, Bt, bias, C, M, Nn, Kp, lda, Kp, ldc, act);
  };
  auto ew = [&](size_t n) { return dim3((unsigned)((n + 255) / 256)); };

  // ---------------- fusion ----------------
  gemm(visual, Dv, wt_fus_wv, fus_bv, T1, d, N, d, Dv, ACT_NONE);            // v
  gemm(audio,  Da, wt_fus_wa, fus_ba, T2, d, N, d, Da, ACT_NONE);            // a
  k_concat2<<<ew((size_t)N * 2 * d), 256, 0, stream>>>(CAT, T1, T2, N, d, d, 0);
  gemm(CAT, 2 * d, wt_fus_wg, fus_bg, T3, d, N, d, 2 * d, ACT_SIGMOID);      // g
  k_gatecomb_ln<<<N, d, 0, stream>>>(nullptr, T1, T2, T3, fus_ln_g, fus_ln_b, X, N, d, 0);

  // ---------------- bidirectional Mamba stack ----------------
  for (int l = 0; l < L; ++l) {
    k_reverse<<<ew((size_t)N * d), 256, 0, stream>>>(XREV, X, N, d);
    for (int dir = 0; dir < 2; ++dir) {
      const int    s   = l * 2 + dir;
      const float* Xin = dir ? XREV : X;
      float*       OUT = dir ? BWD : F;
      gemm(Xin, d, wt_in[s], nullptr, XZ, 2 * di, N, 2 * di, d, ACT_NONE);    // in-proj
      k_conv_silu<<<ew((size_t)N * di), 256, 0, stream>>>(
          XZ, m_conv_w + (size_t)s * di * dc, m_conv_b + (size_t)s * di, U, N, di, dc);
      gemm(U, di, wt_xp[s], nullptr, DBC, 48, N, 48, di, ACT_NONE);           // x-proj
      gemm(DBC, 48, wt_dt[s], m_dt_b + (size_t)s * di,
           DEL, di, N, di, dr, ACT_SOFTPLUS);                                 // dt-proj
      k_scan<<<(di + 255) / 256, 256, 0, stream>>>(
          DBC, DEL, U, m_Alog + (size_t)s * di * 16, m_D + (size_t)s * di, Y, N, di);
      k_ymul<<<ew((size_t)N * di), 256, 0, stream>>>(Y, XZ, N, di);
      gemm(Y, di, wt_out[s], nullptr, OUT, d, N, d, di, ACT_NONE);            // out-proj
    }
    k_concat2<<<ew((size_t)N * 2 * d), 256, 0, stream>>>(CAT, F, BWD, N, d, d, 1);
    gemm(CAT, 2 * d, wt_gw[l], g_b + (size_t)l * d, T3, d, N, d, 2 * d, ACT_SIGMOID);
    k_gatecomb_ln<<<N, d, 0, stream>>>(X, F, BWD, T3,
                                       enc_ln_g + (size_t)l * d, enc_ln_b + (size_t)l * d,
                                       X, N, d, 1);
  }

  // ---------------- multi-scale pooling ----------------
  float* P3 = DEL;  // N x 3d, spans DEL+Y regions (mamba temps now free)
  gemm(X, d, wt_pool[0], pool_b, P3, 3 * d, N, d, d, ACT_NONE);
  k_pool<<<ew((size_t)N * d), 256, 0, stream>>>(X, T2, N, d, 2);
  gemm(T2, d, wt_pool[1], pool_b + d, P3 + d, 3 * d, N, d, d, ACT_NONE);
  k_pool<<<ew((size_t)N * d), 256, 0, stream>>>(X, T2, N, d, 4);
  gemm(T2, d, wt_pool[2], pool_b + 2 * d, P3 + 2 * d, 3 * d, N, d, d, ACT_NONE);
  gemm(P3, 3 * d, wt_comb, comb_b, T3, d, N, d, 3 * d, ACT_RELU);
  k_ln<<<N, d, 0, stream>>>(T3, comb_ln_g, comb_ln_b, T1, d);  // xp -> T1

  // ---------------- checkpoint cross-attention ----------------
  float* Qb = XZ;
  float* Kb = Qb + (size_t)N * d;
  float* Vb = Kb + (size_t)Kc * d;
  float* Ob = Vb + (size_t)Kc * d;
  float* XC = Ob + (size_t)N * d;
  float* HH = XC + (size_t)N * d;
  float* CP = HH + (size_t)N * 128;
  k_gather<<<ew((size_t)Kc * d), 256, 0, stream>>>(CP, T1, cp_idx, Kc, d);
  gemm(T1, d, wt_qkv[0], qkv_b,         Qb, d, N,  d, d, ACT_NONE);
  gemm(CP, d, wt_qkv[1], qkv_b + d,     Kb, d, Kc, d, d, ACT_NONE);
  gemm(CP, d, wt_qkv[2], qkv_b + 2 * d, Vb, d, Kc, d, d, ACT_NONE);
  k_attn<<<N, 128, 0, stream>>>(Qb, Kb, Vb, Ob, Kc);
  gemm(Ob, d, wt_ao, attn_out_b, F, d, N, d, d, ACT_NONE);
  k_res_ln<<<N, d, 0, stream>>>(F, T1, cp_ln_g, cp_ln_b, XC, d);

  // ---------------- scoring head ----------------
  gemm(XC, d, wt_s1, s1_b, HH, 128, N, 128, d, ACT_RELU);
  k_ln<<<N, 128, 0, stream>>>(HH, s_ln_g, s_ln_b, HH, 128);
  k_score<<<(N + 255) / 256, 256, 0, stream>>>(HH, s2_w, s2_b, (float*)d_out, N, 128);

  // encoded output (second tuple element) = X after the Mamba stack
  (void)hipMemcpyAsync((float*)d_out + N, X, (size_t)N * d * sizeof(float),
                       hipMemcpyDeviceToDevice, stream);
}
